// MultiHeadedAttention_64665027608991
// MI455X (gfx1250) — matrix operand order repaired and checked by reading, not compiled, not run on hardware
//
#include <hip/hip_runtime.h>
#include <hip/hip_bf16.h>

#define TIME_LEN 8
#define JOINT 22
#define LSEQ 176          // TIME_LEN * JOINT
#define LT 11             // 176 / 16 m-tiles
#define IN_DIM 128
#define HD 64
#define HN 8
#define DM 512            // HN * HD
#define LPAD 192          // 176 padded to multiple of 32 for P@V

typedef __attribute__((ext_vector_type(16))) _Float16 v16h;
typedef __attribute__((ext_vector_type(8)))  _Float16 v8h;
typedef __attribute__((ext_vector_type(4)))  _Float16 v4h;
typedef __attribute__((ext_vector_type(8)))  float    v8f;

static __device__ __forceinline__ v8f wmma_f16(v16h a, v16h b, v8f c) {
    return __builtin_amdgcn_wmma_f32_16x16x32_f16(false, a, false, b, (short)0, c, false, false);
}

// A-fragment (16x32 f16): lane holds row = base_row + (lane&15).
// elements 0..15 = K (khalf*16 .. khalf*16+15), khalf = lane>>4
static __device__ __forceinline__ v16h load_a_frag(const _Float16* __restrict__ base,
                                                   int stride, int row, int k0, int lane) {
    int khalf = lane >> 4;
    const _Float16* p = base + (size_t)row * stride + k0 + khalf * 16;
    union { v16h v; v8h h[2]; } u;
    u.h[0] = *(const v8h*)(p);
    u.h[1] = *(const v8h*)(p + 8);
    return u.v;
}

// ---------------------------------------------------------------------------
// Kernel 0: convert + transpose weights to f16.  WT[n][k] = W[k][n]
// ---------------------------------------------------------------------------
__global__ void prep_weights(const float* __restrict__ Wq, const float* __restrict__ Wk,
                             const float* __restrict__ Wv, const float* __restrict__ Wf,
                             _Float16* __restrict__ WqT, _Float16* __restrict__ WkT,
                             _Float16* __restrict__ WvT, _Float16* __restrict__ WfT) {
    int i = blockIdx.x * blockDim.x + threadIdx.x;
    if (i < DM * IN_DIM) {                 // 512 x 128
        int n = i >> 7, k = i & 127;
        WqT[i] = (_Float16)Wq[k * DM + n];
        WkT[i] = (_Float16)Wk[k * DM + n];
        WvT[i] = (_Float16)Wv[k * DM + n];
    }
    if (i < DM * DM) {                     // 512 x 512
        int n = i >> 9, k = i & 511;
        WfT[i] = (_Float16)Wf[k * DM + n];
    }
}

// ---------------------------------------------------------------------------
// Kernel 1: fused QKV projection + masked softmax attention per (batch, head).
// Block = 352 threads = 11 waves. Grid = B * 8.
// ---------------------------------------------------------------------------
__launch_bounds__(352, 1)
__global__ void attn_kernel(const float* __restrict__ x,
                            const _Float16* __restrict__ WqT, const float* __restrict__ bq,
                            const _Float16* __restrict__ WkT, const float* __restrict__ bk,
                            const _Float16* __restrict__ WvT, const float* __restrict__ bv,
                            _Float16* __restrict__ attn_out) {
    __shared__ __align__(32) unsigned char sUnion[LT * 16 * LPAD * 2]; // x (45KB) then P (66KB)
    __shared__ __align__(32) _Float16 sQ[LSEQ * HD];      // [l][d] row-major, pre-scaled 1/8
    __shared__ __align__(32) _Float16 sK[LSEQ * HD];      // [l][d] row-major
    __shared__ __align__(32) _Float16 sVt[HD * LPAD];     // TRANSPOSED: [d][l], l padded to 192

    _Float16* sX = (_Float16*)sUnion;   // phase A/B: x[b] as f16  [176 x 128]
    _Float16* sP = (_Float16*)sUnion;   // phase C: per-wave P tiles [11][16 x 192]

    const int tid  = threadIdx.x;
    const int wave = __builtin_amdgcn_readfirstlane(tid >> 5);  // wave-uniform -> SGPR
    const int lane = tid & 31;
    const int khalf = lane >> 4;
    const int lane16 = lane & 15;
    const int b = blockIdx.x >> 3;
    const int h = blockIdx.x & 7;

    // ---- Phase A: stage x[b] into LDS as f16 (vectorized); zero V pad cols --
    const float4* xb4 = (const float4*)(x + (size_t)b * LSEQ * IN_DIM);
    for (int i = tid; i < LSEQ * IN_DIM / 4; i += 352) {
        float4 f = xb4[i];
        v4h hv; hv[0] = (_Float16)f.x; hv[1] = (_Float16)f.y;
                hv[2] = (_Float16)f.z; hv[3] = (_Float16)f.w;
        *(v4h*)(sX + i * 4) = hv;
    }
    // zero sVt[d][176..191] for all 64 d
    for (int i = tid; i < HD * (LPAD - LSEQ) / 8; i += 352) {
        int d = i >> 1, j = (i & 1) * 8;
        *(v8h*)(sVt + d * LPAD + LSEQ + j) = (v8h){};
    }
    __syncthreads();

    // ---- Phase B: Q/K/V projections via WMMA. 12 strips over 11 waves ----
    for (int s = wave; s < 12; s += 11) {
        const int mat = s >> 2;        // 0=Q 1=K 2=V  (scalar)
        const int nt  = s & 3;         // n-tile within head dim (scalar)
        const _Float16* Wt = (mat == 0) ? WqT : (mat == 1) ? WkT : WvT;
        const float* bias  = (mat == 0) ? bq  : (mat == 1) ? bk  : bv;

        const int ncol = h * HD + nt * 16 + lane16;   // global output column
        // preload the 4 B-fragments of this column strip (K = 128)
        v16h Bf[4];
#pragma unroll
        for (int kc = 0; kc < 4; ++kc)
            Bf[kc] = *(const v16h*)(Wt + (size_t)ncol * IN_DIM + kc * 32 + khalf * 16);
        const float bval = bias[ncol];

        if (mat == 2) {
            // V: ReLU, store TRANSPOSED sVt[d][l]; 8 consecutive l -> one b128
            const int col = nt * 16 + lane16;                // d
            for (int m = 0; m < LT; ++m) {
                v8f acc = {};
#pragma unroll
                for (int kc = 0; kc < 4; ++kc) {
                    v16h Af = load_a_frag(sX, IN_DIM, m * 16 + lane16, kc * 32, lane);
                    acc = wmma_f16(Af, Bf[kc], acc);
                }
                v8h pk;
#pragma unroll
                for (int e = 0; e < 8; ++e)
                    pk[e] = (_Float16)fmaxf(acc[e] + bval, 0.f);
                *(v8h*)(sVt + col * LPAD + m * 16 + 8 * khalf) = pk;
            }
        } else {
            _Float16* dst = (mat == 0) ? sQ : sK;
            const float postscale = (mat == 0) ? 0.125f : 1.0f;  // bake 1/sqrt(64) into Q
            for (int m = 0; m < LT; ++m) {
                v8f acc = {};
#pragma unroll
                for (int kc = 0; kc < 4; ++kc) {
                    v16h Af = load_a_frag(sX, IN_DIM, m * 16 + lane16, kc * 32, lane);
                    acc = wmma_f16(Af, Bf[kc], acc);
                }
#pragma unroll
                for (int e = 0; e < 8; ++e) {
                    int row = m * 16 + e + 8 * khalf;
                    dst[row * HD + nt * 16 + lane16] = (_Float16)((acc[e] + bval) * postscale);
                }
            }
        }
    }
    __syncthreads();

    // ---- Phase C: attention. wave w owns query tile w (rows w*16..w*16+15) --
    const int mrow = wave * 16 + lane16;

    // per-element query row index / frame (independent of nt -> hoisted)
    int rowIdx[8], rframe[8];
#pragma unroll
    for (int e = 0; e < 8; ++e) {
        rowIdx[e] = wave * 16 + e + 8 * khalf;
        rframe[e] = rowIdx[e] / JOINT;
    }

    v16h Qa[2];
#pragma unroll
    for (int kc = 0; kc < 2; ++kc)
        Qa[kc] = load_a_frag(sQ, HD, mrow, kc * 32, lane);

    // scores S = (Q/8) K^T with temporal mask (scale already in Q)
    v8f S[LT];
#pragma unroll
    for (int nt = 0; nt < LT; ++nt) {
        v8f acc = {};
#pragma unroll
        for (int kc = 0; kc < 2; ++kc) {
            v16h Bf = *(const v16h*)(sK + (nt * 16 + lane16) * HD + kc * 32 + khalf * 16);
            acc = wmma_f16(Qa[kc], Bf, acc);
        }
        const int col = nt * 16 + lane16;          // key index
        const int cframe = col / JOINT;
#pragma unroll
        for (int e = 0; e < 8; ++e) {
            bool keep = (rframe[e] != cframe) || (rowIdx[e] == col);
            acc[e] = keep ? acc[e] : -1.0e30f;
        }
        S[nt] = acc;
    }

    // row softmax: rows live across the 16 lanes of each half-wave
#pragma unroll
    for (int e = 0; e < 8; ++e) {
        float mx = -3.0e38f;
#pragma unroll
        for (int nt = 0; nt < LT; ++nt) mx = fmaxf(mx, S[nt][e]);
#pragma unroll
        for (int off = 8; off >= 1; off >>= 1) mx = fmaxf(mx, __shfl_xor(mx, off, 16));
        float sum = 0.f;
#pragma unroll
        for (int nt = 0; nt < LT; ++nt) { float p = __expf(S[nt][e] - mx); S[nt][e] = p; sum += p; }
#pragma unroll
        for (int off = 8; off >= 1; off >>= 1) sum += __shfl_xor(sum, off, 16);
        float inv = 1.0f / sum;
#pragma unroll
        for (int nt = 0; nt < LT; ++nt) S[nt][e] *= inv;
    }

    // stage P tile (16 x 192, zero padded cols) into per-wave LDS region
    _Float16* sPw = sP + wave * 16 * LPAD;
#pragma unroll
    for (int nt = 0; nt < LT; ++nt)
#pragma unroll
        for (int e = 0; e < 8; ++e)
            sPw[(e + 8 * khalf) * LPAD + nt * 16 + lane16] = (_Float16)S[nt][e];
    for (int r = khalf; r < 16; r += 2)
        sPw[r * LPAD + LSEQ + lane16] = (_Float16)0.f;
    __syncthreads();

    // O = P @ V : [16 x 192] x [192 x 64]; V fragments are contiguous in sVt
    v8f O[4] = {{}, {}, {}, {}};
#pragma unroll
    for (int kc = 0; kc < 6; ++kc) {
        v16h Pa = load_a_frag(sPw, LPAD, lane16, kc * 32, lane);
#pragma unroll
        for (int nt2 = 0; nt2 < 4; ++nt2) {
            v16h Bf = *(const v16h*)(sVt + (nt2 * 16 + lane16) * LPAD + kc * 32 + khalf * 16);
            O[nt2] = wmma_f16(Pa, Bf, O[nt2]);
        }
    }

    // store head output as f16: attn_out[b][l][h*64 + d]
#pragma unroll
    for (int nt2 = 0; nt2 < 4; ++nt2) {
        const int col = h * HD + nt2 * 16 + lane16;
#pragma unroll
        for (int e = 0; e < 8; ++e) {
            int row = wave * 16 + e + 8 * khalf;
            attn_out[((size_t)b * LSEQ + row) * DM + col] = (_Float16)O[nt2][e];
        }
    }
}

// ---------------------------------------------------------------------------
// Kernel 2: final projection  out[M x 512] = attn[M x 512] @ Wf + bf
// Each wave: 16(M) x 64(N) strip, K = 512 in 16 chunks.
// ---------------------------------------------------------------------------
__launch_bounds__(256, 2)
__global__ void gemm_kernel(const _Float16* __restrict__ A,   // [M x 512] f16
                            const _Float16* __restrict__ WfT, // [512 x 512] f16 (transposed)
                            const float* __restrict__ bf,
                            float* __restrict__ out, int mtiles) {
    const int wave = __builtin_amdgcn_readfirstlane(threadIdx.x >> 5);
    const int lane = threadIdx.x & 31;
    const int khalf = lane >> 4;
    const int lane16 = lane & 15;
    const int mtile = blockIdx.x * 8 + wave;
    if (mtile >= mtiles) return;
    const int nbase = blockIdx.y * 64;

    v8f acc[4] = {{}, {}, {}, {}};
#pragma unroll 4
    for (int kc = 0; kc < 16; ++kc) {
        v16h Af = load_a_frag(A, DM, mtile * 16 + lane16, kc * 32, lane);
#pragma unroll
        for (int j = 0; j < 4; ++j) {
            const int n = nbase + j * 16 + lane16;
            v16h Bf = *(const v16h*)(WfT + (size_t)n * DM + kc * 32 + khalf * 16);
            acc[j] = wmma_f16(Af, Bf, acc[j]);
        }
    }
#pragma unroll
    for (int j = 0; j < 4; ++j) {
        const int n = nbase + j * 16 + lane16;
        const float bval = bf[n];
#pragma unroll
        for (int e = 0; e < 8; ++e) {
            int row = mtile * 16 + e + 8 * khalf;
            out[(size_t)row * DM + n] = acc[j][e] + bval;
        }
    }
}

// ---------------------------------------------------------------------------
extern "C" void kernel_launch(void* const* d_in, const int* in_sizes, int n_in,
                              void* d_out, int out_size, void* d_ws, size_t ws_size,
                              hipStream_t stream) {
    const float* x  = (const float*)d_in[0];
    const float* Wq = (const float*)d_in[1];
    const float* bq = (const float*)d_in[2];
    const float* Wk = (const float*)d_in[3];
    const float* bk = (const float*)d_in[4];
    const float* Wv = (const float*)d_in[5];
    const float* bv = (const float*)d_in[6];
    const float* Wf = (const float*)d_in[7];
    const float* bf = (const float*)d_in[8];
    float* out = (float*)d_out;

    const int B = in_sizes[0] / (LSEQ * IN_DIM);

    _Float16* WqT  = (_Float16*)d_ws;
    _Float16* WkT  = WqT + DM * IN_DIM;
    _Float16* WvT  = WkT + DM * IN_DIM;
    _Float16* WfT  = WvT + DM * IN_DIM;
    _Float16* attn = WfT + DM * DM;       // [B * 176 x 512] f16

    prep_weights<<<(DM * DM + 255) / 256, 256, 0, stream>>>(Wq, Wk, Wv, Wf,
                                                            WqT, WkT, WvT, WfT);

    attn_kernel<<<B * HN, 352, 0, stream>>>(x, WqT, bq, WkT, bk, WvT, bv, attn);

    const int mtiles = B * LT;            // (B*176)/16
    dim3 g2((mtiles + 7) / 8, DM / 64);
    gemm_kernel<<<g2, 256, 0, stream>>>(attn, WfT, bf, out, mtiles);
}
